// MultiHeadAttention_6107443495349
// MI455X (gfx1250) — matrix operand order repaired and checked by reading, not compiled, not run on hardware
//
#include <hip/hip_runtime.h>
#include <math.h>

// MI455X / gfx1250, wave32. B=8, S=1024, D=768, H=12, DK=64.

typedef __attribute__((ext_vector_type(16))) _Float16 v16h;
typedef __attribute__((ext_vector_type(8)))  float    v8f;

#define Bn 8
#define Sn 1024
#define Dn 768
#define Hn 12
#define DKn 64

union FragH {
    v16h v;
    _Float16 h[16];
    unsigned int u[8];
    uint4 q[2];
};

#define ASTRIDE 36   // A tile row stride in f32 (144B: 16B-aligned chunks, conflict-free)
#define BSTRIDE 20   // B tile row stride in f32 (80B: 16B-aligned chunks, conflict-free)

// ---------------------------------------------------------------------------
// Kernel 1: Y16[b,h,s,dk] (f16) = X[(b,s),:] @ W + bias, head-major layout.
// Block: 128 threads (4 waves). Block tile: 64 rows x 16 cols. K step 32.
// Tiles stream into double-buffered LDS via GLOBAL_LOAD_ASYNC_TO_LDS_B128
// (ASYNCcnt), overlapping the next tile's DMA with the current WMMA.
// ---------------------------------------------------------------------------
__global__ __launch_bounds__(128)
void proj_gemm_f16(const float* __restrict__ X, const float* __restrict__ W,
                   const float* __restrict__ bias, _Float16* __restrict__ Y16) {
    __shared__ __align__(16) float Abuf[2][64 * ASTRIDE];   // 2 x 9216 B
    __shared__ __align__(16) float Bbuf[2][32 * BSTRIDE];   // 2 x 2560 B

    const int tid  = threadIdx.x;
    const int wave = tid >> 5;
    const int lane = tid & 31;
    const int n0   = blockIdx.x * 16;    // output col tile (48 tiles)
    const int m0   = blockIdx.y * 64;    // output row tile (128 tiles)

    // Issue async DMA of one K-tile (A: 64x32 f32, B: 32x16 f32) into buffer `buf`.
    auto issue_tile = [&](int buf, int k0) {
        // A: 512 chunks of 16B; thread t covers chunks t, t+128, t+256, t+384.
#pragma unroll
        for (int i = 0; i < 4; ++i) {
            const int chunk = tid + 128 * i;
            const int row   = chunk >> 3;        // 0..63
            const int cc    = chunk & 7;         // 4-float group within row
            const float* g = X + (size_t)(m0 + row) * Dn + k0 + cc * 4;
            unsigned lds = (unsigned)(unsigned long long)&Abuf[buf][row * ASTRIDE + cc * 4];
            unsigned long long ga = (unsigned long long)g;
            asm volatile("global_load_async_to_lds_b128 %0, %1, off"
                         :: "v"(lds), "v"(ga) : "memory");
        }
        // B: 128 chunks of 16B; one per thread.
        {
            const int row = tid >> 2;            // 0..31 (k)
            const int cc  = tid & 3;             // 4-float group within row (n)
            const float* g = W + (size_t)(k0 + row) * Dn + n0 + cc * 4;
            unsigned lds = (unsigned)(unsigned long long)&Bbuf[buf][row * BSTRIDE + cc * 4];
            unsigned long long ga = (unsigned long long)g;
            asm volatile("global_load_async_to_lds_b128 %0, %1, off"
                         :: "v"(lds), "v"(ga) : "memory");
        }
    };

    v8f c = {};
    const int mrow = wave * 16 + (lane & 15);
    const int hh   = lane >> 4;
    const int ncol = lane & 15;

    issue_tile(0, 0);
    asm volatile("s_wait_asynccnt 0x0" ::: "memory");
    __syncthreads();

    int cur = 0;
    for (int k0 = 0; k0 < Dn; k0 += 32) {
        if (k0 + 32 < Dn) issue_tile(cur ^ 1, k0 + 32);   // prefetch next tile

        // Build f16 fragments from f32 LDS (contiguous K = 16h+e for both A and B).
        FragH a, b;
        const float* arow = &Abuf[cur][mrow * ASTRIDE + 16 * hh];
#pragma unroll
        for (int e = 0; e < 8; ++e)  a.h[e] = (_Float16)arow[e];        // K = 16h+e
#pragma unroll
        for (int e = 8; e < 16; ++e) a.h[e] = (_Float16)arow[e];        // K = 16h+e
        const float* bcol = &Bbuf[cur][16 * hh * BSTRIDE + ncol];
#pragma unroll
        for (int e = 0; e < 16; ++e) b.h[e] = (_Float16)bcol[e * BSTRIDE]; // K = 16h+e

        c = __builtin_amdgcn_wmma_f32_16x16x32_f16(false, a.v, false, b.v,
                                                   (short)0, c, false, false);

        // All waves done reading `cur`; next buffer's DMA complete.
        asm volatile("s_wait_asynccnt 0x0" ::: "memory");
        __syncthreads();
        cur ^= 1;
    }

    // Epilogue: + bias, convert to f16, scatter to [B,H,S,DK] layout.
    const int Ng = n0 + ncol;
    const float bv = bias[Ng];
    const int head = Ng >> 6;
    const int dk   = Ng & 63;
#pragma unroll
    for (int v = 0; v < 8; ++v) {
        const int Mg = m0 + wave * 16 + v + 8 * hh;
        const int bb = Mg >> 10;
        const int s  = Mg & 1023;
        Y16[((size_t)(bb * Hn + head) * Sn + s) * DKn + dk] = (_Float16)(c[v] + bv);
    }
}

// ---------------------------------------------------------------------------
// Kernel 2: asp[b,dk] = aspect[b,:] @ Wd + bd   (tiny)
// ---------------------------------------------------------------------------
__global__ __launch_bounds__(256)
void asp_proj(const float* __restrict__ aspect, const float* __restrict__ Wd,
              const float* __restrict__ bd, float* __restrict__ asp) {
    const int t = blockIdx.x * 256 + threadIdx.x;
    if (t >= Bn * DKn) return;
    const int bb = t >> 6, dk = t & 63;
    float acc = bd[dk];
    const float* av = aspect + bb * Dn;
    for (int d = 0; d < Dn; ++d) acc += av[d] * Wd[d * DKn + dk];
    asp[t] = acc;
}

// ---------------------------------------------------------------------------
// Kernel 3: ascore[b,h,j] = tanh(asp[b,:] . k16[b,h,j,:] + bias_m)
// ---------------------------------------------------------------------------
__global__ __launch_bounds__(256)
void asp_scores_k(const _Float16* __restrict__ k16, const float* __restrict__ asp,
                  const float* __restrict__ bias_m, float* __restrict__ ascore) {
    const int g  = blockIdx.x * 256 + threadIdx.x;   // B*H*S = 98304
    const int bh = g >> 10;
    const int bb = bh / Hn;
    const _Float16* kr = k16 + (size_t)g * DKn;
    const float* av = asp + bb * DKn;
    float acc = bias_m[0];
#pragma unroll
    for (int d = 0; d < DKn; ++d) acc += av[d] * (float)kr[d];
    ascore[g] = tanhf(acc);
}

// ---------------------------------------------------------------------------
// Kernel 4: scores = softmax( mask(qk/8 + aspect) + short ) over j.
// Block: 256 threads (8 waves), tile = 16 rows x 1024 cols of one (b,h).
// Each wave: 8 j-tiles of 16 cols; 2 chained WMMAs (K=64) per tile.
// Row block staged in 64KB LDS, then wave-cooperative softmax (wave32).
// ---------------------------------------------------------------------------
__global__ __launch_bounds__(256)
void attn_scores(const _Float16* __restrict__ q16, const _Float16* __restrict__ k16,
                 const float* __restrict__ ascore, const int* __restrict__ maskp,
                 const float* __restrict__ shortp, float* __restrict__ out) {
    __shared__ float sc[16 * Sn];        // 64 KB

    const int i0   = blockIdx.x * 16;
    const int bh   = blockIdx.y;
    const int bb   = bh / Hn;
    const int wave = threadIdx.x >> 5;
    const int lane = threadIdx.x & 31;
    const int n    = lane & 15;
    const int hh   = lane >> 4;

    // A fragments: q rows i0..i0+15 for this (b,h); 16B-aligned uint4 chunks.
    FragH a0, a1;
    {
        const uint4* qrow = (const uint4*)(q16 + ((size_t)bh * Sn + i0 + n) * DKn);
        a0.q[0] = qrow[hh];          // K 0..31, e<8
        a0.q[1] = qrow[2 + hh];      // K 0..31, e>=8
        a1.q[0] = qrow[4 + hh];      // K 32..63
        a1.q[1] = qrow[6 + hh];
    }

    for (int t = 0; t < 8; ++t) {
        const int j0 = wave * 128 + t * 16;
        // B fragments: column n of k^T = row (j0+n) of k16, 16B chunks in table K order.
        FragH b0, b1;
        {
            const uint4* krow = (const uint4*)(k16 + ((size_t)bh * Sn + j0 + n) * DKn);
            b0.q[0] = krow[hh];
            b0.q[1] = krow[2 + hh];
            b1.q[0] = krow[4 + hh];
            b1.q[1] = krow[6 + hh];
        }
        v8f c = {};
        c = __builtin_amdgcn_wmma_f32_16x16x32_f16(false, a0.v, false, b0.v,
                                                   (short)0, c, false, false);
        c = __builtin_amdgcn_wmma_f32_16x16x32_f16(false, a1.v, false, b1.v,
                                                   (short)0, c, false, false);

        const int j = j0 + n;
        const float aspv = ascore[(size_t)bh * Sn + j];
        const int   mk   = maskp[bb * Sn + j];
        const float* srow = shortp + ((size_t)bh * Sn + i0) * Sn + j;
        if (t + 1 < 8) __builtin_prefetch(srow + 16, 0, 1);   // next short tile
#pragma unroll
        for (int v = 0; v < 8; ++v) {
            const int m = v + 8 * hh;
            float s = c[v] * 0.125f + aspv;          // /sqrt(64) + aspect bcast
            if (mk == 0) s = -1e9f;                  // mask BEFORE adding short
            s += srow[(size_t)m * Sn];               // + short[b,h,i,j]
            sc[m * Sn + j] = s;
        }
    }
    __syncthreads();

    // Softmax over j: 16 rows / 8 waves = 2 rows per wave (wave32 reductions).
    for (int rr = 0; rr < 2; ++rr) {
        const int r = wave * 2 + rr;
        const float* row = sc + r * Sn;
        float mx = -3.4e38f;
        for (int jj = lane; jj < Sn; jj += 32) mx = fmaxf(mx, row[jj]);
#pragma unroll
        for (int o = 16; o > 0; o >>= 1) mx = fmaxf(mx, __shfl_xor(mx, o, 32));
        float sum = 0.0f;
        for (int jj = lane; jj < Sn; jj += 32) sum += __expf(row[jj] - mx);
#pragma unroll
        for (int o = 16; o > 0; o >>= 1) sum += __shfl_xor(sum, o, 32);
        const float inv = 1.0f / sum;
        float* orow = out + ((size_t)bh * Sn + i0 + r) * Sn;
        for (int jj = lane; jj < Sn; jj += 32)
            orow[jj] = __expf(row[jj] - mx) * inv;
    }
}

// ---------------------------------------------------------------------------
extern "C" void kernel_launch(void* const* d_in, const int* in_sizes, int n_in,
                              void* d_out, int out_size, void* d_ws, size_t ws_size,
                              hipStream_t stream) {
    (void)in_sizes; (void)n_in; (void)out_size; (void)ws_size;
    const float* query  = (const float*)d_in[0];
    const float* key_   = (const float*)d_in[1];
    const float* shortp = (const float*)d_in[2];
    const float* aspect = (const float*)d_in[3];
    const int*   maskp  = (const int*)  d_in[4];
    const float* Wq     = (const float*)d_in[5];
    const float* bq     = (const float*)d_in[6];
    const float* Wk     = (const float*)d_in[7];
    const float* bk     = (const float*)d_in[8];
    const float* Wd     = (const float*)d_in[9];
    const float* bd     = (const float*)d_in[10];
    const float* bias_m = (const float*)d_in[11];
    float* out = (float*)d_out;

    // workspace layout (all 16B-aligned)
    const size_t q16_bytes = (size_t)Bn * Hn * Sn * DKn * 2;   // 12,582,912
    char* ws = (char*)d_ws;
    _Float16* q16   = (_Float16*)ws;
    _Float16* k16   = (_Float16*)(ws + q16_bytes);
    float*    asp   = (float*)(ws + 2 * q16_bytes);                   // 2 KB
    float*    ascr  = (float*)(ws + 2 * q16_bytes + 2048);            // 384 KB

    dim3 gproj(Dn / 16, (Bn * Sn) / 64);          // 48 x 128
    proj_gemm_f16<<<gproj, 128, 0, stream>>>(query, Wq, bq, q16);
    proj_gemm_f16<<<gproj, 128, 0, stream>>>(key_, Wk, bk, k16);
    asp_proj<<<2, 256, 0, stream>>>(aspect, Wd, bd, asp);
    asp_scores_k<<<(Bn * Hn * Sn) / 256, 256, 0, stream>>>(k16, asp, bias_m, ascr);
    attn_scores<<<dim3(Sn / 16, Bn * Hn), 256, 0, stream>>>(q16, k16, ascr, maskp,
                                                            shortp, out);
}
